// BertModel_17059610100479
// MI455X (gfx1250) — matrix operand order repaired and checked by reading, not compiled, not run on hardware
//
#include <hip/hip_runtime.h>
#include <hip/hip_bf16.h>
#include <cstdint>

// ---------------------------------------------------------------------------
// BERT-base forward (B=8, S=512, H=768, L=12, NH=12, D=64, I=3072) on gfx1250.
// ~770 GFLOP vs <1 GB traffic => compute-bound => every GEMM runs on
// v_wmma_f32_16x16x32_bf16 (wave32 WMMA, f32 accumulate).
// Weights are convert+transposed once per launch to bf16 W^T[N][K] so all GEMM
// B tiles (and A tiles) are staged with GLOBAL_LOAD_ASYNC_TO_LDS_B128 into
// double-buffered LDS (ASYNCcnt pipeline, one barrier per K-step).
// ---------------------------------------------------------------------------

typedef __bf16 bf16;
typedef __attribute__((ext_vector_type(16))) __bf16 v16bf;
typedef __attribute__((ext_vector_type(8)))  __bf16 v8bf;
typedef __attribute__((ext_vector_type(8)))  float  v8f;

static constexpr int Bn  = 8;
static constexpr int Sn  = 512;
static constexpr int Hn  = 768;
static constexpr int Ln  = 12;
static constexpr int NHn = 12;
static constexpr int Dn  = 64;
static constexpr int In  = 3072;
static constexpr int Mtot = Bn * Sn;          // 4096 token rows

// Block tile: 128 (M) x 64 (N), K-step 32. 8 waves, each owns a 32x32 tile.
#define BM 128
#define BN 64
#define BK 32

__device__ __forceinline__ v16bf mk16(v8bf lo, v8bf hi) {
  v16bf r;
#pragma unroll
  for (int i = 0; i < 8; ++i) { r[i] = lo[i]; r[i + 8] = hi[i]; }
  return r;
}

// 16-byte async global->LDS copy (ASYNCcnt-tracked).  vdst = LDS byte offset,
// vaddr = 64-bit global address, saddr = off.  (CDNA5 ISA)
__device__ __forceinline__ void async_cp_b128(unsigned lds_off, const void* g) {
  asm volatile("global_load_async_to_lds_b128 %0, %1, off"
               :: "v"(lds_off), "v"((unsigned long long)(uintptr_t)g)
               : "memory");
}

__device__ __forceinline__ void wait_async0() {
#if __has_builtin(__builtin_amdgcn_s_wait_asynccnt)
  __builtin_amdgcn_s_wait_asynccnt(0);
#else
  asm volatile("s_wait_asynccnt 0x0" ::: "memory");
#endif
}

__device__ __forceinline__ unsigned lds_off(const void* p) {
  return (unsigned)(uintptr_t)p;   // low 32 bits of generic ptr = LDS offset
}

// ---------------------------------------------------------------------------
// WMMA GEMM:  C[z][M][N] = act( A[z] @ B[z] + bias ) + res (+ scale/mask bias)
// BMODE 1: B = bf16 [N,K] (transposed weights, and K-matrix for QK^T) - async
// BMODE 2: B = bf16 [K,N] (PV: V rows are [key][d]; transpose during stage)
// LDS B tile stored [N][K] so each lane's fragment is 2 contiguous 16B loads.
// Double-buffered LDS: stage tile kt+1 (async) before computing tile kt.
// ---------------------------------------------------------------------------
template <int BMODE>
__global__ __launch_bounds__(256)
void wgemm_kernel(const bf16* __restrict__ A, int64_t aOut, int64_t aIn, int lda,
                  const bf16* __restrict__ Bsrc, int64_t bOut, int64_t bIn, int ldb,
                  const float* __restrict__ bias,
                  const float* __restrict__ Res, int ldr,
                  float* __restrict__ Cf, int64_t cfOut, int64_t cfIn, int ldcf,
                  bf16* __restrict__ Cb, int64_t cbOut, int64_t cbIn, int ldcb,
                  const int* __restrict__ mask, float scale, int act,
                  int zdiv, int N, int K)
{
  __shared__ bf16 As[2][BM][BK];   // 16 KB (double-buffered)
  __shared__ bf16 Bs[2][BN][BK];   //  8 KB, transposed: Bs[.][n][k]

  const int tid  = threadIdx.x;
  const int lane = tid & 31;
  const int l16  = lane & 15;
  const int hl   = lane >> 4;           // half-wave select
  const int wave = tid >> 5;            // 0..7
  const int m0   = (wave >> 1) * 32;
  const int n0   = (wave & 1)  * 32;

  const int z     = blockIdx.z;
  const int zo    = z / zdiv;
  const int zi    = z % zdiv;
  const int m_blk = blockIdx.y * BM;
  const int n_blk = blockIdx.x * BN;

  const bf16* Ap = A    + zo * aOut + zi * aIn;
  const bf16* Bp = Bsrc + zo * bOut + zi * bIn;

  v8f acc[2][2];
#pragma unroll
  for (int i = 0; i < 2; ++i)
#pragma unroll
    for (int j = 0; j < 2; ++j)
#pragma unroll
      for (int r = 0; r < 8; ++r) acc[i][j][r] = 0.0f;

  auto stage = [&](int buf, int k0) {
    {   // A tile 128x32 bf16: 2 x async b128 per thread
      const int row = tid >> 1;
      const int c0  = (tid & 1) * 16;
      const bf16* src = Ap + (int64_t)(m_blk + row) * lda + k0 + c0;
      async_cp_b128(lds_off(&As[buf][row][c0]),     src);
      async_cp_b128(lds_off(&As[buf][row][c0 + 8]), src + 8);
    }
    if constexpr (BMODE == 1) {       // B [N,K] -> direct async copy
      const int nn = tid >> 2;
      const int c0 = (tid & 3) * 8;
      const bf16* src = Bp + (int64_t)(n_blk + nn) * ldb + k0 + c0;
      async_cp_b128(lds_off(&Bs[buf][nn][c0]), src);
    } else {                          // B [K,N] -> transpose during stage
      const int kk = tid >> 3;
      const int nn = (tid & 7) * 8;
      const bf16* src = Bp + (int64_t)(k0 + kk) * ldb + n_blk + nn;
      v8bf v = *(const v8bf*)src;
#pragma unroll
      for (int i = 0; i < 8; ++i) Bs[buf][nn + i][kk] = v[i];
    }
  };

  const int ksteps = K / BK;
  stage(0, 0);
  for (int kt = 0; kt < ksteps; ++kt) {
    wait_async0();
    __syncthreads();                         // tile kt ready for all waves
    if (kt + 1 < ksteps) stage((kt + 1) & 1, (kt + 1) * BK);

    const int buf = kt & 1;
    v16bf af[2], bfg[2];
#pragma unroll
    for (int tm = 0; tm < 2; ++tm) {         // A frag: lanes 16-31 start K=8
      const int row = m0 + tm * 16 + l16;
      const int kb  = hl * 8;
      v8bf lo = *(const v8bf*)&As[buf][row][kb];
      v8bf hi = *(const v8bf*)&As[buf][row][kb + 16];
      af[tm] = mk16(lo, hi);
    }
#pragma unroll
    for (int tn = 0; tn < 2; ++tn) {         // B frag: lanes 16-31 start K=8
      const int nr = n0 + tn * 16 + l16;
      const int kb = hl * 8;
      v8bf lo = *(const v8bf*)&Bs[buf][nr][kb];
      v8bf hi = *(const v8bf*)&Bs[buf][nr][kb + 16];
      bfg[tn] = mk16(lo, hi);
    }
#pragma unroll
    for (int tm = 0; tm < 2; ++tm)
#pragma unroll
      for (int tn = 0; tn < 2; ++tn)
        acc[tm][tn] = __builtin_amdgcn_wmma_f32_16x16x32_bf16(
            false, af[tm], false, bfg[tn], (short)0, acc[tm][tn], false, false);
  }

  // ---- epilogue: bias / scale+mask-bias / GELU / residual / dual store ----
  const int64_t cfoff = Cf ? (zo * cfOut + zi * cfIn) : 0;
  const int64_t cboff = Cb ? (zo * cbOut + zi * cbIn) : 0;
#pragma unroll
  for (int tm = 0; tm < 2; ++tm)
#pragma unroll
    for (int tn = 0; tn < 2; ++tn) {
#pragma unroll
      for (int r = 0; r < 8; ++r) {
        const int Mrow = m_blk + m0 + tm * 16 + hl * 8 + r;
        const int Ncol = n_blk + n0 + tn * 16 + l16;
        float val = acc[tm][tn][r];
        if (bias) val += bias[Ncol];
        if (mask) {
          const float mb = (1.0f - (float)mask[(int64_t)zo * N + Ncol]) * -10000.0f;
          val = val * scale + mb;
        }
        if (act == 1)                               // exact GELU
          val = 0.5f * val * (1.0f + erff(val * 0.70710678118654752f));
        if (Res) val += Res[(int64_t)Mrow * ldr + Ncol];
        if (Cf) Cf[cfoff + (int64_t)Mrow * ldcf + Ncol] = val;
        if (Cb) Cb[cboff + (int64_t)Mrow * ldcb + Ncol] = (bf16)val;
      }
    }
}

// ---------------------------------------------------------------------------
// Weight convert + transpose: fp32 W[K,N] -> bf16 WT[N,K]  (32x32 LDS tiles)
// ---------------------------------------------------------------------------
__global__ __launch_bounds__(256)
void convT_kernel(const float* __restrict__ W, bf16* __restrict__ WT,
                  int Kd, int Nd)
{
  __shared__ bf16 tile[32][33];
  const int tx = threadIdx.x & 31;
  const int ty = threadIdx.x >> 5;         // 0..7
  const int n0 = blockIdx.x * 32;
  const int k0 = blockIdx.y * 32;
#pragma unroll
  for (int i = 0; i < 4; ++i)
    tile[ty + i * 8][tx] = (bf16)W[(int64_t)(k0 + ty + i * 8) * Nd + n0 + tx];
  __syncthreads();
#pragma unroll
  for (int i = 0; i < 4; ++i)
    WT[(int64_t)(n0 + ty + i * 8) * Kd + k0 + tx] = tile[tx][ty + i * 8];
}

// ---------------------------------------------------------------------------
// Embedding gather: tok + type + pos
// ---------------------------------------------------------------------------
__global__ __launch_bounds__(256)
void emb_kernel(const int* __restrict__ ids, const int* __restrict__ tt,
                const float* __restrict__ tok, const float* __restrict__ typ,
                const float* __restrict__ pos, float* __restrict__ out)
{
  const int m = blockIdx.x;
  const int t = threadIdx.x;
  const int s = m & (Sn - 1);
  const int id = ids[m];
  const int ty = tt[m];
#pragma unroll
  for (int i = 0; i < 3; ++i) {
    const int j = t + i * 256;
    out[(int64_t)m * Hn + j] =
        tok[(int64_t)id * Hn + j] + typ[(int64_t)ty * Hn + j] + pos[(int64_t)s * Hn + j];
  }
}

// ---------------------------------------------------------------------------
// LayerNorm (H=768): fp32 + bf16 outputs
// ---------------------------------------------------------------------------
__global__ __launch_bounds__(256)
void ln_kernel(const float* __restrict__ x, const float* __restrict__ g,
               const float* __restrict__ b, float* __restrict__ of,
               bf16* __restrict__ ob)
{
  const int row = blockIdx.x;
  const int t   = threadIdx.x;
  const float* xr = x + (int64_t)row * Hn;
  float v0 = xr[t], v1 = xr[t + 256], v2 = xr[t + 512];

  __shared__ float red[256];
  red[t] = v0 + v1 + v2;
  __syncthreads();
  for (int off = 128; off > 0; off >>= 1) {
    if (t < off) red[t] += red[t + off];
    __syncthreads();
  }
  const float mu = red[0] * (1.0f / (float)Hn);
  __syncthreads();

  const float d0 = v0 - mu, d1 = v1 - mu, d2 = v2 - mu;
  red[t] = d0 * d0 + d1 * d1 + d2 * d2;
  __syncthreads();
  for (int off = 128; off > 0; off >>= 1) {
    if (t < off) red[t] += red[t + off];
    __syncthreads();
  }
  const float rstd = rsqrtf(red[0] * (1.0f / (float)Hn) + 1e-3f);

  const float y0 = d0 * rstd * g[t]       + b[t];
  const float y1 = d1 * rstd * g[t + 256] + b[t + 256];
  const float y2 = d2 * rstd * g[t + 512] + b[t + 512];
  float* orow = of + (int64_t)row * Hn;
  orow[t] = y0; orow[t + 256] = y1; orow[t + 512] = y2;
  bf16* brow = ob + (int64_t)row * Hn;
  brow[t] = (bf16)y0; brow[t + 256] = (bf16)y1; brow[t + 512] = (bf16)y2;
}

// ---------------------------------------------------------------------------
// Softmax over S=512 keys -> bf16 probs
// ---------------------------------------------------------------------------
__global__ __launch_bounds__(256)
void softmax_kernel(const float* __restrict__ s, bf16* __restrict__ p)
{
  const int row = blockIdx.x;
  const int t   = threadIdx.x;
  const float* x = s + (int64_t)row * Sn;
  float a = x[t], b = x[t + 256];

  __shared__ float red[256];
  red[t] = fmaxf(a, b);
  __syncthreads();
  for (int off = 128; off > 0; off >>= 1) {
    if (t < off) red[t] = fmaxf(red[t], red[t + off]);
    __syncthreads();
  }
  const float mx = red[0];
  __syncthreads();

  const float ea = expf(a - mx), eb = expf(b - mx);
  red[t] = ea + eb;
  __syncthreads();
  for (int off = 128; off > 0; off >>= 1) {
    if (t < off) red[t] += red[t + off];
    __syncthreads();
  }
  const float inv = 1.0f / red[0];
  bf16* pr = p + (int64_t)row * Sn;
  pr[t] = (bf16)(ea * inv);
  pr[t + 256] = (bf16)(eb * inv);
}

// ---------------------------------------------------------------------------
// Host orchestration
// ---------------------------------------------------------------------------
extern "C" void kernel_launch(void* const* d_in, const int* in_sizes, int n_in,
                              void* d_out, int out_size, void* d_ws, size_t ws_size,
                              hipStream_t stream)
{
  (void)in_sizes; (void)n_in; (void)out_size; (void)ws_size;

  const int*   input_ids  = (const int*)d_in[0];
  const int*   input_mask = (const int*)d_in[1];
  const int*   token_type = (const int*)d_in[2];
  const float* tok_emb    = (const float*)d_in[3];
  const float* pos_emb    = (const float*)d_in[4];
  const float* type_emb   = (const float*)d_in[5];
  const float* emb_g      = (const float*)d_in[6];
  const float* emb_b      = (const float*)d_in[7];
  const float* Wq = (const float*)d_in[8];   const float* bq = (const float*)d_in[9];
  const float* Wk = (const float*)d_in[10];  const float* bk = (const float*)d_in[11];
  const float* Wv = (const float*)d_in[12];  const float* bv = (const float*)d_in[13];
  const float* Wo = (const float*)d_in[14];  const float* bo = (const float*)d_in[15];
  const float* g1 = (const float*)d_in[16];  const float* b1 = (const float*)d_in[17];
  const float* Wi = (const float*)d_in[18];  const float* bi = (const float*)d_in[19];
  const float* Wd = (const float*)d_in[20];  const float* bd = (const float*)d_in[21];
  const float* g2 = (const float*)d_in[22];  const float* b2 = (const float*)d_in[23];

  // ---- carve workspace ----
  char* w = (char*)d_ws;
  auto carve = [&](size_t bytes) { char* p = w; w += (bytes + 255) & ~size_t(255); return p; };
  float* x_f     = (float*)carve((size_t)Mtot * Hn * 4);
  bf16*  xb      = (bf16*) carve((size_t)Mtot * Hn * 2);
  float* t0_f    = (float*)carve((size_t)Mtot * Hn * 4);
  float* attn_f  = (float*)carve((size_t)Mtot * Hn * 4);
  bf16*  attnb   = (bf16*) carve((size_t)Mtot * Hn * 2);
  bf16*  q_b     = (bf16*) carve((size_t)Mtot * Hn * 2);
  bf16*  k_b     = (bf16*) carve((size_t)Mtot * Hn * 2);
  bf16*  v_b     = (bf16*) carve((size_t)Mtot * Hn * 2);
  bf16*  o_b     = (bf16*) carve((size_t)Mtot * Hn * 2);
  bf16*  ffn1_b  = (bf16*) carve((size_t)Mtot * In * 2);
  float* scores  = (float*)carve((size_t)Bn * NHn * Sn * Sn * 4);
  bf16*  probs   = (bf16*) carve((size_t)Bn * NHn * Sn * Sn * 2);
  const size_t HH = (size_t)Hn * Hn, HI = (size_t)Hn * In;
  const size_t perLayerT = 4 * HH + 2 * HI;                 // bf16 elements
  bf16* wT = (bf16*)carve(Ln * perLayerT * 2);              // transposed weights

  const dim3 blk(256);
  const dim3 gH(Hn / BN, Mtot / BM, 1);              // 12 x 32  (N=768 GEMMs)
  const dim3 gI(In / BN, Mtot / BM, 1);              // 48 x 32  (N=3072 GEMM)
  const dim3 gS(Sn / BN, Sn / BM, Bn * NHn);         // 8 x 4 x 96 (scores)
  const dim3 gPV(Dn / BN, Sn / BM, Bn * NHn);        // 1 x 4 x 96 (probs @ V)
  const int64_t SH = (int64_t)Sn * Hn;
  const int64_t SS = (int64_t)Sn * Sn;

  // ---- one-shot weight convert+transpose: fp32 W[K,N] -> bf16 WT[N,K] ----
  for (int l = 0; l < Ln; ++l) {
    bf16* base = wT + (size_t)l * perLayerT;
    convT_kernel<<<dim3(Hn/32, Hn/32), blk, 0, stream>>>(Wq + (size_t)l*HH, base,          Hn, Hn);
    convT_kernel<<<dim3(Hn/32, Hn/32), blk, 0, stream>>>(Wk + (size_t)l*HH, base + HH,     Hn, Hn);
    convT_kernel<<<dim3(Hn/32, Hn/32), blk, 0, stream>>>(Wv + (size_t)l*HH, base + 2*HH,   Hn, Hn);
    convT_kernel<<<dim3(Hn/32, Hn/32), blk, 0, stream>>>(Wo + (size_t)l*HH, base + 3*HH,   Hn, Hn);
    convT_kernel<<<dim3(In/32, Hn/32), blk, 0, stream>>>(Wi + (size_t)l*HI, base + 4*HH,   Hn, In);
    convT_kernel<<<dim3(Hn/32, In/32), blk, 0, stream>>>(Wd + (size_t)l*HI, base + 4*HH + HI, In, Hn);
  }

  // ---- embeddings + LN ----
  emb_kernel<<<Mtot, blk, 0, stream>>>(input_ids, token_type, tok_emb, type_emb,
                                       pos_emb, t0_f);
  ln_kernel<<<Mtot, blk, 0, stream>>>(t0_f, emb_g, emb_b, x_f, xb);

  for (int l = 0; l < Ln; ++l) {
    bf16* base = wT + (size_t)l * perLayerT;
    const bf16 *wqT = base, *wkT = base + HH, *wvT = base + 2*HH, *woT = base + 3*HH;
    const bf16 *wiT = base + 4*HH, *wdT = base + 4*HH + HI;
    const float* bql = bq + l * Hn;  const float* bkl = bk + l * Hn;
    const float* bvl = bv + l * Hn;  const float* bol = bo + l * Hn;
    const float* bil = bi + l * In;  const float* bdl = bd + l * Hn;
    const float* g1l = g1 + l * Hn;  const float* b1l = b1 + l * Hn;
    const float* g2l = g2 + l * Hn;  const float* b2l = b2 + l * Hn;

    // Q/K/V projections: x @ W + b  -> bf16 [B*S, H]
    wgemm_kernel<1><<<gH, blk, 0, stream>>>(xb, 0, 0, Hn, wqT, 0, 0, Hn, bql,
        nullptr, 0, nullptr, 0, 0, 0, q_b, 0, 0, Hn, nullptr, 1.0f, 0, 1, Hn, Hn);
    wgemm_kernel<1><<<gH, blk, 0, stream>>>(xb, 0, 0, Hn, wkT, 0, 0, Hn, bkl,
        nullptr, 0, nullptr, 0, 0, 0, k_b, 0, 0, Hn, nullptr, 1.0f, 0, 1, Hn, Hn);
    wgemm_kernel<1><<<gH, blk, 0, stream>>>(xb, 0, 0, Hn, wvT, 0, 0, Hn, bvl,
        nullptr, 0, nullptr, 0, 0, 0, v_b, 0, 0, Hn, nullptr, 1.0f, 0, 1, Hn, Hn);

    // scores[b,h] = (Q K^T) * 1/sqrt(D) + mask_bias   (K rows already B^T)
    wgemm_kernel<1><<<gS, blk, 0, stream>>>(q_b, SH, Dn, Hn,
        k_b, SH, Dn, Hn, nullptr, nullptr, 0,
        scores, (int64_t)NHn * SS, SS, Sn,
        nullptr, 0, 0, 0,
        input_mask, 0.125f, 0, NHn, Sn, Dn);

    softmax_kernel<<<Bn * NHn * Sn, blk, 0, stream>>>(scores, probs);

    // O[b,h] = probs @ V  -> merged heads bf16 [B*S, H]
    wgemm_kernel<2><<<gPV, blk, 0, stream>>>(probs, (int64_t)NHn * SS, SS, Sn,
        v_b, SH, Dn, Hn, nullptr, nullptr, 0,
        nullptr, 0, 0, 0, o_b, SH, Dn, Hn,
        nullptr, 1.0f, 0, NHn, Dn, Sn);

    // attn_pre = O @ Wo + bo + x (residual)
    wgemm_kernel<1><<<gH, blk, 0, stream>>>(o_b, 0, 0, Hn, woT, 0, 0, Hn, bol,
        x_f, Hn, t0_f, 0, 0, Hn, nullptr, 0, 0, 0, nullptr, 1.0f, 0, 1, Hn, Hn);
    ln_kernel<<<Mtot, blk, 0, stream>>>(t0_f, g1l, b1l, attn_f, attnb);

    // FFN1 = GELU(attn @ Wi + bi)
    wgemm_kernel<1><<<gI, blk, 0, stream>>>(attnb, 0, 0, Hn, wiT, 0, 0, Hn, bil,
        nullptr, 0, nullptr, 0, 0, 0, ffn1_b, 0, 0, In, nullptr, 1.0f, 1, 1, In, Hn);

    // FFN2 = ffn1 @ Wd + bd + attn (residual)
    wgemm_kernel<1><<<gH, blk, 0, stream>>>(ffn1_b, 0, 0, In, wdT, 0, 0, In, bdl,
        attn_f, Hn, t0_f, 0, 0, Hn, nullptr, 0, 0, 0, nullptr, 1.0f, 0, 1, Hn, In);

    float* lnout = (l == Ln - 1) ? (float*)d_out : x_f;
    ln_kernel<<<Mtot, blk, 0, stream>>>(t0_f, g2l, b2l, lnout, xb);
  }
}
